// GNNModel_1898375545384
// MI455X (gfx1250) — matrix operand order repaired and checked by reading, not compiled, not run on hardware
//
#include <hip/hip_runtime.h>
#include <hip/hip_bf16.h>
#include <math.h>

// ---------------------------------------------------------------------------
// Model dimensions (from the reference)
// ---------------------------------------------------------------------------
#define HIDDEN   512
#define VDIM     8
#define NGRAPH   512
#define NNODES   10240
#define NPATHS   5
#define SEQLEN   9
#define BJ       (NGRAPH * NPATHS)    // 2560 sequences
#define NTOK     (BJ * SEQLEN)        // 23040 tokens
#define LDIM     64
#define FFNDIM   2048
#define FEATK    835                  // 1 + 1 + 1 + 512 + 320
#define FEATP    864                  // FEATK padded to a multiple of 32

typedef __attribute__((ext_vector_type(16))) _Float16 v16h;
typedef __attribute__((ext_vector_type(8)))  float    v8f;
typedef __attribute__((ext_vector_type(4)))  int      v4i;

// ---------------------------------------------------------------------------
// Device helpers
// ---------------------------------------------------------------------------
__device__ __forceinline__ float siluf(float x) { return x / (1.f + __expf(-x)); }
__device__ __forceinline__ float actf(float x, int act) {
  if (act == 1) return fmaxf(x, 0.f);
  if (act == 2) return siluf(x);
  return x;
}
__device__ __forceinline__ float wave_sum32(float v) {
#pragma unroll
  for (int o = 16; o > 0; o >>= 1) v += __shfl_xor(v, o, 32);
  return v;
}

// Pack two 8-float runs (at kb and kb+16) into the 16-half A fragment,
// per the CDNA5 ISA 16-bit A 16x32 lane layout.
__device__ __forceinline__ v16h pack_a(const float4 a0, const float4 a1,
                                       const float4 a2, const float4 a3) {
  v16h a;
  a[0]  = (_Float16)a0.x; a[1]  = (_Float16)a0.y; a[2]  = (_Float16)a0.z; a[3]  = (_Float16)a0.w;
  a[4]  = (_Float16)a1.x; a[5]  = (_Float16)a1.y; a[6]  = (_Float16)a1.z; a[7]  = (_Float16)a1.w;
  a[8]  = (_Float16)a2.x; a[9]  = (_Float16)a2.y; a[10] = (_Float16)a2.z; a[11] = (_Float16)a2.w;
  a[12] = (_Float16)a3.x; a[13] = (_Float16)a3.y; a[14] = (_Float16)a3.z; a[15] = (_Float16)a3.w;
  return a;
}

// ---------------------------------------------------------------------------
// Stage a 32x64 f32 B tile (rows kt..kt+31, cols n0..n0+63 of W with row
// stride N) into LDS. Uses gfx1250 async global->LDS b128 copies (ASYNCcnt)
// when the builtins exist, else coalesced float4 loads + LDS stores.
// Each wave waits for its own async copies; all waves then barrier (here)
// before reading the tile.
// ---------------------------------------------------------------------------
#if defined(__gfx1250__) && __has_builtin(__builtin_amdgcn_global_load_async_to_lds_b128)
#define USE_ASYNC_LDS 1
#else
#define USE_ASYNC_LDS 0
#endif

__device__ __forceinline__ void stage_b_tile(const float* __restrict__ W,
                                             int kt, int N, int n0, int tid,
                                             float (*Bt)[64]) {
#if USE_ASYNC_LDS
#pragma unroll
  for (int j = 0; j < 4; ++j) {
    int f = tid + j * 128;
    int r = f >> 4;
    int c4 = (f & 15) << 2;
    const float* gp = W + (size_t)(kt + r) * N + n0 + c4;
    float* lp = &Bt[r][c4];
    __builtin_amdgcn_global_load_async_to_lds_b128(
        (__attribute__((address_space(1))) v4i*)gp,
        (__attribute__((address_space(3))) v4i*)lp, 0, 0);
  }
#if __has_builtin(__builtin_amdgcn_s_wait_asynccnt)
  __builtin_amdgcn_s_wait_asynccnt(0);
#else
  asm volatile("s_wait_asynccnt 0x0" ::: "memory");
#endif
#else
#pragma unroll
  for (int j = 0; j < 4; ++j) {
    int f = tid + j * 128;
    int r = f >> 4;
    int c4 = (f & 15) << 2;
    float4 w4 = *reinterpret_cast<const float4*>(W + (size_t)(kt + r) * N + n0 + c4);
    Bt[r][c4 + 0] = w4.x; Bt[r][c4 + 1] = w4.y;
    Bt[r][c4 + 2] = w4.z; Bt[r][c4 + 3] = w4.w;
  }
#endif
  __syncthreads();
}

// ---------------------------------------------------------------------------
// WMMA GEMM: C(M,N) = act(A(M,K; row stride lda) @ W(K,N) + bias)
// Block = 128 threads = 4 waves. Block tile = 64Mx64N; wave w owns rows
// m0+16w..+15 and computes 4 col sub-tiles. B tile (32x64) staged in LDS.
// Requires: M%64==0, N%64==0, K%32==0, lda%4==0.
// ---------------------------------------------------------------------------
__global__ void gemm_wmma_kernel(const float* __restrict__ A, int lda,
                                 const float* __restrict__ W,
                                 const float* __restrict__ bias,
                                 float* __restrict__ C,
                                 int K, int N, int act) {
  __shared__ float Bt[32][64];
  const int tid  = threadIdx.x;
  const int lane = tid & 31;
  const int wave = tid >> 5;
  const int n0 = blockIdx.x * 64;
  const int m0 = blockIdx.y * 64 + wave * 16;

  const float* Arow = A + (size_t)(m0 + (lane & 15)) * lda;
  const int colLocal = lane & 15;
  const int kb16 = (lane & 16) ? 16 : 0;

  v8f acc[4] = {};

  for (int kt = 0; kt < K; kt += 32) {
    stage_b_tile(W, kt, N, n0, tid, Bt);

    const int kb = kt + ((lane & 16) ? 8 : 0);
    __builtin_prefetch(Arow + kt + 128, 0, 1);
    float4 a0 = *reinterpret_cast<const float4*>(Arow + kb);
    float4 a1 = *reinterpret_cast<const float4*>(Arow + kb + 4);
    float4 a2 = *reinterpret_cast<const float4*>(Arow + kb + 16);
    float4 a3 = *reinterpret_cast<const float4*>(Arow + kb + 20);
    v16h a = pack_a(a0, a1, a2, a3);

#pragma unroll
    for (int s = 0; s < 4; ++s) {
      int col = s * 16 + colLocal;
      v16h b;
#pragma unroll
      for (int i = 0; i < 16; ++i) b[i] = (_Float16)Bt[(kb16 >> 1) + i + ((i >> 3) << 3)][col];
      acc[s] = __builtin_amdgcn_wmma_f32_16x16x32_f16(false, a, false, b,
                                                      (short)0, acc[s], false, false);
    }
    __syncthreads();
  }

  const int rbase = m0 + ((lane & 16) ? 8 : 0);
#pragma unroll
  for (int s = 0; s < 4; ++s) {
    int col = n0 + s * 16 + colLocal;
    float bb = bias ? bias[col] : 0.f;
#pragma unroll
    for (int i = 0; i < 8; ++i) {
      C[(size_t)(rbase + i) * N + col] = actf(acc[s][i] + bb, act);
    }
  }
}

// ---------------------------------------------------------------------------
// Fused norm-GEMM: Out(NNODES, C) = sqrt( sum_j ( (gate.*V[n,j,:]) @ W[:,c] )^2 )
// V is (NNODES, VDIM, K). Block = 128 threads; tile = 16 nodes x 64 cols,
// wave w owns col sub-tile w. One B fragment per kt reused across all VDIM
// planes (8 accumulators); gate vector loaded once per kt.
// Requires: C%64==0, K%32==0.
// ---------------------------------------------------------------------------
__global__ void vnorm_wmma_kernel(const float* __restrict__ V,
                                  const float* __restrict__ W,
                                  const float* __restrict__ gateBase,
                                  int gStride, int gOff,
                                  float* __restrict__ Out, int K, int C) {
  __shared__ float Bt[32][64];
  const int tid  = threadIdx.x;
  const int lane = tid & 31;
  const int wave = tid >> 5;
  const int n0 = blockIdx.x * 64;
  const int m0 = blockIdx.y * 16;

  const int row = m0 + (lane & 15);    // node index for this lane's A rows
  const int colLocal = wave * 16 + (lane & 15);
  const int kb16 = (lane & 16) ? 16 : 0;
  const float* grow = gateBase ? gateBase + (size_t)row * gStride + gOff : nullptr;

  v8f acc[VDIM] = {};

  for (int kt = 0; kt < K; kt += 32) {
    stage_b_tile(W, kt, C, n0, tid, Bt);

    const int kb = kt + ((lane & 16) ? 8 : 0);

    // B fragment for this wave's col tile (shared across all VDIM planes)
    v16h b;
#pragma unroll
    for (int i = 0; i < 16; ++i) b[i] = (_Float16)Bt[(kb16 >> 1) + i + ((i >> 3) << 3)][colLocal];

    // gate values for this lane's k positions (shared across planes)
    float4 g0 = make_float4(1.f, 1.f, 1.f, 1.f), g1 = g0, g2 = g0, g3 = g0;
    if (gateBase) {
      g0 = *reinterpret_cast<const float4*>(grow + kb);
      g1 = *reinterpret_cast<const float4*>(grow + kb + 4);
      g2 = *reinterpret_cast<const float4*>(grow + kb + 16);
      g3 = *reinterpret_cast<const float4*>(grow + kb + 20);
    }

#pragma unroll
    for (int j = 0; j < VDIM; ++j) {
      const float* Arow = V + ((size_t)row * VDIM + j) * K;
      float4 a0 = *reinterpret_cast<const float4*>(Arow + kb);
      float4 a1 = *reinterpret_cast<const float4*>(Arow + kb + 4);
      float4 a2 = *reinterpret_cast<const float4*>(Arow + kb + 16);
      float4 a3 = *reinterpret_cast<const float4*>(Arow + kb + 20);
      a0.x *= g0.x; a0.y *= g0.y; a0.z *= g0.z; a0.w *= g0.w;
      a1.x *= g1.x; a1.y *= g1.y; a1.z *= g1.z; a1.w *= g1.w;
      a2.x *= g2.x; a2.y *= g2.y; a2.z *= g2.z; a2.w *= g2.w;
      a3.x *= g3.x; a3.y *= g3.y; a3.z *= g3.z; a3.w *= g3.w;
      v16h a = pack_a(a0, a1, a2, a3);
      acc[j] = __builtin_amdgcn_wmma_f32_16x16x32_f16(false, a, false, b,
                                                      (short)0, acc[j], false, false);
    }
    __syncthreads();
  }

  const int rbase = m0 + ((lane & 16) ? 8 : 0);
  const int col = n0 + colLocal;
#pragma unroll
  for (int i = 0; i < 8; ++i) {
    float s = 0.f;
#pragma unroll
    for (int j = 0; j < VDIM; ++j) s += acc[j][i] * acc[j][i];
    Out[(size_t)(rbase + i) * C + col] = sqrtf(s);
  }
}

// ---------------------------------------------------------------------------
// Attention (S=9 tiny -> VALU). Thread per (seq, head, q).
// ---------------------------------------------------------------------------
template <int DMODEL, int NH, bool MASKED>
__global__ void attn_kernel(const float* __restrict__ qkv,
                            const unsigned char* __restrict__ mask,
                            float* __restrict__ out) {
  constexpr int DH = DMODEL / NH;
  constexpr int QW = 3 * DMODEL;
  int idx = blockIdx.x * blockDim.x + threadIdx.x;
  if (idx >= BJ * NH * SEQLEN) return;
  int q = idx % SEQLEN;
  int h = (idx / SEQLEN) % NH;
  int s = idx / (SEQLEN * NH);

  const float* base = qkv + (size_t)s * SEQLEN * QW;
  const float* qp = base + (size_t)q * QW + h * DH;
  const float scale = rsqrtf((float)DH);

  float sc[SEQLEN];
  float mx = -3.0e38f;
  for (int k = 0; k < SEQLEN; ++k) {
    const float* kp = base + (size_t)k * QW + DMODEL + h * DH;
    float acc = 0.f;
    for (int d = 0; d < DH; ++d) acc += qp[d] * kp[d];
    acc *= scale;
    if (MASKED && !mask[(size_t)s * SEQLEN + k]) acc = -1.0e9f;
    sc[k] = acc;
    mx = fmaxf(mx, acc);
  }
  float den = 0.f;
  for (int k = 0; k < SEQLEN; ++k) { sc[k] = __expf(sc[k] - mx); den += sc[k]; }
  float inv = 1.f / den;

  float* op = out + ((size_t)s * SEQLEN + q) * DMODEL + h * DH;
  for (int d = 0; d < DH; ++d) {
    float acc = 0.f;
    for (int k = 0; k < SEQLEN; ++k)
      acc += sc[k] * base[(size_t)k * QW + 2 * DMODEL + h * DH + d];
    op[d] = acc * inv;
  }
}

// ---------------------------------------------------------------------------
// Residual + LayerNorm, wave per token: x = LN(x + f) * g + b
// ---------------------------------------------------------------------------
__global__ void add_ln_kernel(float* __restrict__ x, const float* __restrict__ f,
                              const float* __restrict__ g, const float* __restrict__ b,
                              int D, int ntok) {
  int wave = (blockIdx.x * blockDim.x + threadIdx.x) >> 5;
  int lane = threadIdx.x & 31;
  if (wave >= ntok) return;
  float* row = x + (size_t)wave * D;
  const float* fr = f + (size_t)wave * D;
  int per = D / 32;   // 16 for D=512, 2 for D=64
  float vals[16];
  float s = 0.f;
  for (int i = 0; i < per; ++i) {
    float v = row[lane + i * 32] + fr[lane + i * 32];
    vals[i] = v;
    s += v;
  }
  s = wave_sum32(s);
  float mean = s / (float)D;
  float vs = 0.f;
  for (int i = 0; i < per; ++i) { float d = vals[i] - mean; vs += d * d; }
  vs = wave_sum32(vs);
  float inv = rsqrtf(vs / (float)D + 1e-5f);
  for (int i = 0; i < per; ++i) {
    int c = lane + i * 32;
    row[c] = (vals[i] - mean) * inv * g[c] + b[c];
  }
}

// ---------------------------------------------------------------------------
// Small mat-vec: out[m] = bias[col] + A[m,:lda] . W[:,col] (W row stride ldw)
// ---------------------------------------------------------------------------
__global__ void matvec_kernel(const float* __restrict__ A, int lda,
                              const float* __restrict__ W, int ldw, int col,
                              const float* __restrict__ bias,
                              float* __restrict__ out, int M, int K) {
  int m = blockIdx.x * blockDim.x + threadIdx.x;
  if (m >= M) return;
  const float* a = A + (size_t)m * lda;
  float acc = bias ? bias[col] : 0.f;
  for (int k = 0; k < K; ++k) acc += a[k] * W[(size_t)k * ldw + col];
  out[m] = acc;
}

// ---------------------------------------------------------------------------
// Glue kernels
// ---------------------------------------------------------------------------
__global__ void concat1_kernel(const float* __restrict__ x,
                               const float* __restrict__ nvm,
                               float* __restrict__ cat) {
  int idx = blockIdx.x * blockDim.x + threadIdx.x;
  if (idx >= NNODES * 1024) return;
  int n = idx >> 10, c = idx & 1023;
  cat[idx] = (c < 512) ? x[(size_t)n * 512 + c] : nvm[(size_t)n * 512 + (c - 512)];
}

__global__ void cat2_kernel(const float* __restrict__ hdd,
                            const float* __restrict__ nvm2,
                            float* __restrict__ cat) {
  int idx = blockIdx.x * blockDim.x + threadIdx.x;
  if (idx >= NNODES * 512) return;
  int n = idx >> 9, c = idx & 511;
  cat[idx] = (c < 256) ? siluf(hdd[(size_t)n * 512 + c])
                       : nvm2[(size_t)n * 256 + (c - 256)];
}

__global__ void seg_sum_kernel(const float* __restrict__ xpre,
                               const int* __restrict__ batch,
                               float* __restrict__ xsum) {
  int i = blockIdx.x * blockDim.x + threadIdx.x;
  if (i >= NNODES) return;
  atomicAdd(&xsum[batch[i]], xpre[i]);
}

__global__ void add_pos_kernel(float* __restrict__ x, const float* __restrict__ pos,
                               int D) {
  int idx = blockIdx.x * blockDim.x + threadIdx.x;
  if (idx >= NTOK * D) return;
  int t = idx / D, c = idx % D;
  x[idx] += pos[(t % SEQLEN) * D + c];
}

__global__ void build_xl_kernel(const float* __restrict__ lab,
                                const float* __restrict__ emb_w,
                                const float* __restrict__ emb_b,
                                const float* __restrict__ pos,
                                float* __restrict__ xl) {
  int idx = blockIdx.x * blockDim.x + threadIdx.x;
  if (idx >= NTOK * LDIM) return;
  int t = idx / LDIM, c = idx % LDIM;
  xl[idx] = lab[t] * emb_w[c] + emb_b[c] + pos[(t % SEQLEN) * LDIM + c];
}

__global__ void pf_mean_kernel(const float* __restrict__ pfseq,
                               float* __restrict__ pf) {
  int idx = blockIdx.x * blockDim.x + threadIdx.x;
  if (idx >= NGRAPH * 512) return;
  int b = idx >> 9, c = idx & 511;
  float acc = 0.f;
  for (int j = 0; j < NPATHS; ++j) acc += pfseq[((size_t)b * NPATHS + j) * 512 + c];
  pf[idx] = acc * (1.f / (float)NPATHS);
}

__global__ void lf_build_kernel(const float* __restrict__ lfseq,
                                const unsigned char* __restrict__ mask,
                                float* __restrict__ lf) {
  int idx = blockIdx.x * blockDim.x + threadIdx.x;
  if (idx >= NGRAPH * NPATHS * LDIM) return;
  int c = idx % LDIM;
  int sj = idx / LDIM;           // b*NPATHS + j
  bool valid = false;
  for (int s = 0; s < SEQLEN; ++s) valid = valid || (mask[(size_t)sj * SEQLEN + s] != 0);
  lf[idx] = valid ? lfseq[(size_t)sj * LDIM + c] : 0.f;
}

__global__ void feat_build_kernel(const float* __restrict__ xs1,
                                  const float* __restrict__ xs2,
                                  const float* __restrict__ pf,
                                  const float* __restrict__ lf,
                                  float* __restrict__ feat) {
  int idx = blockIdx.x * blockDim.x + threadIdx.x;
  if (idx >= NGRAPH * FEATP) return;
  int b = idx / FEATP, c = idx % FEATP;
  float v;
  if (c == 0)         v = xs1[b] - xs2[b];
  else if (c == 1)    v = xs1[b];
  else if (c == 2)    v = xs2[b];
  else if (c < 515)   v = pf[(size_t)b * 512 + (c - 3)];
  else if (c < FEATK) v = lf[(size_t)b * (NPATHS * LDIM) + (c - 515)];
  else                v = 0.f;    // zero pad 835..863
  feat[idx] = v;
}

// zero-pad head w1 (FEATK x 512) into (FEATP x 512)
__global__ void padw_kernel(const float* __restrict__ w, float* __restrict__ wp) {
  int idx = blockIdx.x * blockDim.x + threadIdx.x;
  if (idx >= FEATP * 512) return;
  int r = idx >> 9;
  wp[idx] = (r < FEATK) ? w[idx] : 0.f;
}

__global__ void out_write_kernel(const float* __restrict__ xone,
                                 const float* __restrict__ xs1,
                                 float* __restrict__ out) {
  int b = blockIdx.x * blockDim.x + threadIdx.x;
  if (b >= NGRAPH) return;
  out[(size_t)b * 2 + 0] = xone[b];
  out[(size_t)b * 2 + 1] = xs1[b];
}

// ---------------------------------------------------------------------------
// Host-side helpers
// ---------------------------------------------------------------------------
static inline void launch_gemm(hipStream_t st, const float* A, int lda,
                               const float* W, const float* bias, float* C,
                               int M, int K, int N, int act) {
  dim3 grid(N / 64, M / 64);
  gemm_wmma_kernel<<<grid, 128, 0, st>>>(A, lda, W, bias, C, K, N, act);
}
static inline void launch_vnorm(hipStream_t st, const float* V, const float* W,
                                const float* gate, int gStride, int gOff,
                                float* Out, int K, int C) {
  dim3 grid(C / 64, NNODES / 16);
  vnorm_wmma_kernel<<<grid, 128, 0, st>>>(V, W, gate, gStride, gOff, Out, K, C);
}
static inline void launch_ew(int count, dim3& g) { g = dim3((count + 255) / 256); }

struct EncLayer {
  const float *b1, *b2, *bo, *bqkv, *ln1_b, *ln1_g, *ln2_b, *ln2_g, *w1, *w2, *wo, *wqkv;
};

// ---------------------------------------------------------------------------
// kernel_launch
//
// Input index map (setup_inputs insertion order; nested dicts flattened in
// jax pytree order = sorted keys; layer lists in order):
//  0 x1  1 v1  2 x2  3 v2  4 paths  5 paths_labels
//  6..17   pre_params  : block1{b1,b2,vec1,vec2,w1,w2} block2{...}
//  18..46  path_params : emb_b, emb_w, fc_b, fc_w, layers[0](12), layers[1](12), pos
//  47..75  label_params: same shape with D=64
//  76..81  head_params : b1,b2,b3,w1,w2,w3
//  82 batch1  83 batch2  84 paths_masks (bool bytes)
// ---------------------------------------------------------------------------
extern "C" void kernel_launch(void* const* d_in, const int* in_sizes, int n_in,
                              void* d_out, int out_size, void* d_ws, size_t ws_size,
                              hipStream_t stream) {
  const float* x1 = (const float*)d_in[0];
  const float* v1 = (const float*)d_in[1];
  const float* x2 = (const float*)d_in[2];
  const float* v2 = (const float*)d_in[3];
  const float* paths = (const float*)d_in[4];
  const float* labels = (const float*)d_in[5];

  // pre_params
  const float* b1_b1   = (const float*)d_in[6];
  const float* b1_b2   = (const float*)d_in[7];
  const float* b1_vec1 = (const float*)d_in[8];
  const float* b1_vec2 = (const float*)d_in[9];
  const float* b1_w1   = (const float*)d_in[10];
  const float* b1_w2   = (const float*)d_in[11];
  const float* b2_b1   = (const float*)d_in[12];
  const float* b2_b2   = (const float*)d_in[13];
  const float* b2_vec1 = (const float*)d_in[14];
  const float* b2_w1   = (const float*)d_in[16];
  const float* b2_w2   = (const float*)d_in[17];
  // (b2.vec2 = d_in[15] feeds only the unused vector output of block2)

  // path encoder params
  const float* p_emb_b = (const float*)d_in[18];
  const float* p_emb_w = (const float*)d_in[19];
  const float* p_fc_b  = (const float*)d_in[20];
  const float* p_fc_w  = (const float*)d_in[21];
  EncLayer pL[2];
  for (int l = 0; l < 2; ++l) {
    int o = 22 + 12 * l;
    pL[l].b1 = (const float*)d_in[o+0];  pL[l].b2 = (const float*)d_in[o+1];
    pL[l].bo = (const float*)d_in[o+2];  pL[l].bqkv = (const float*)d_in[o+3];
    pL[l].ln1_b = (const float*)d_in[o+4]; pL[l].ln1_g = (const float*)d_in[o+5];
    pL[l].ln2_b = (const float*)d_in[o+6]; pL[l].ln2_g = (const float*)d_in[o+7];
    pL[l].w1 = (const float*)d_in[o+8];  pL[l].w2 = (const float*)d_in[o+9];
    pL[l].wo = (const float*)d_in[o+10]; pL[l].wqkv = (const float*)d_in[o+11];
  }
  const float* p_pos = (const float*)d_in[46];

  // label encoder params
  const float* l_emb_b = (const float*)d_in[47];
  const float* l_emb_w = (const float*)d_in[48];
  const float* l_fc_b  = (const float*)d_in[49];
  const float* l_fc_w  = (const float*)d_in[50];
  EncLayer lL[2];
  for (int l = 0; l < 2; ++l) {
    int o = 51 + 12 * l;
    lL[l].b1 = (const float*)d_in[o+0];  lL[l].b2 = (const float*)d_in[o+1];
    lL[l].bo = (const float*)d_in[o+2];  lL[l].bqkv = (const float*)d_in[o+3];
    lL[l].ln1_b = (const float*)d_in[o+4]; lL[l].ln1_g = (const float*)d_in[o+5];
    lL[l].ln2_b = (const float*)d_in[o+6]; lL[l].ln2_g = (const float*)d_in[o+7];
    lL[l].w1 = (const float*)d_in[o+8];  lL[l].w2 = (const float*)d_in[o+9];
    lL[l].wo = (const float*)d_in[o+10]; lL[l].wqkv = (const float*)d_in[o+11];
  }
  const float* l_pos = (const float*)d_in[75];

  // head
  const float* h_b1 = (const float*)d_in[76];
  const float* h_b2 = (const float*)d_in[77];
  const float* h_b3 = (const float*)d_in[78];
  const float* h_w1 = (const float*)d_in[79];
  const float* h_w2 = (const float*)d_in[80];
  const float* h_w3 = (const float*)d_in[81];

  const int* batch1 = (const int*)d_in[82];
  const int* batch2 = (const int*)d_in[83];
  const unsigned char* masks = (const unsigned char*)d_in[84];

  float* out = (float*)d_out;

  // ---- workspace layout (floats). Persistent first, then a reusable arena.
  float* ws = (float*)d_ws;
  size_t off = 0;
  float* xsum1 = ws + off; off += NGRAPH;
  float* xsum2 = ws + off; off += NGRAPH;
  float* pf    = ws + off; off += (size_t)NGRAPH * 512;
  float* lf    = ws + off; off += (size_t)NGRAPH * NPATHS * LDIM;
  float* xone  = ws + off; off += NGRAPH;
  off = (off + 63) & ~(size_t)63;                       // keep regions 256B aligned
  float* w1p   = ws + off; off += (size_t)FEATP * 512;  // padded head.w1
  float* arena = ws + off;

  (void)hipMemsetAsync(xsum1, 0, 2 * NGRAPH * sizeof(float), stream);

  dim3 g;
  launch_ew(FEATP * 512, g);
  padw_kernel<<<g, 256, 0, stream>>>(h_w1, w1p);

  // =========================================================================
  // Phase A: pre_reduce + segment sum (per node set)
  // =========================================================================
  for (int set = 0; set < 2; ++set) {
    const float* X = set ? x2 : x1;
    const float* V = set ? v2 : v1;
    const int* batch = set ? batch2 : batch1;
    float* xs = set ? xsum2 : xsum1;

    float* nvm1   = arena;                                   // NNODES*512
    float* vec2b1 = nvm1   + (size_t)NNODES * 512;           // NNODES*8*256
    float* cat1   = vec2b1 + (size_t)NNODES * VDIM * 256;    // NNODES*1024
    float* h1     = cat1   + (size_t)NNODES * 1024;          // NNODES*512
    float* hdd    = h1     + (size_t)NNODES * 512;           // NNODES*512
    float* nvm2   = hdd    + (size_t)NNODES * 512;           // NNODES*256
    float* cat2   = nvm2   + (size_t)NNODES * 256;           // NNODES*512
    float* h2     = cat2   + (size_t)NNODES * 512;           // NNODES*256
    float* xpre   = h2     + (size_t)NNODES * 256;           // NNODES

    // block1: vec1 norm, vec2, MLP
    launch_vnorm(stream, V, b1_vec1, nullptr, 0, 0, nvm1, 512, 512);
    launch_gemm(stream, V, 512, b1_vec2, nullptr, vec2b1, NNODES * VDIM, 512, 256, 0);
    launch_ew(NNODES * 1024, g);
    concat1_kernel<<<g, 256, 0, stream>>>(X, nvm1, cat1);
    launch_gemm(stream, cat1, 1024, b1_w1, b1_b1, h1, NNODES, 1024, 512, 2);  // silu
    launch_gemm(stream, h1, 512, b1_w2, b1_b2, hdd, NNODES, 512, 512, 0);
    // hdd[:, :256] -> xo (silu later), hdd[:, 256:] -> gate

    // block2: vec1 norm with fused gate (vo = vec2b1 * gate), MLP, x output col 0
    launch_vnorm(stream, vec2b1, b2_vec1, hdd, 512, 256, nvm2, 256, 256);
    launch_ew(NNODES * 512, g);
    cat2_kernel<<<g, 256, 0, stream>>>(hdd, nvm2, cat2);
    launch_gemm(stream, cat2, 512, b2_w1, b2_b1, h2, NNODES, 512, 256, 2);    // silu
    launch_ew(NNODES, g);
    matvec_kernel<<<g, 256, 0, stream>>>(h2, 256, b2_w2, 2, 0, b2_b2, xpre, NNODES, 256);
    seg_sum_kernel<<<g, 256, 0, stream>>>(xpre, batch, xs);
  }

  // =========================================================================
  // Phase B: path transformer encoder (D=512, 4 heads, FFN=2048)
  // =========================================================================
  {
    float* xenc  = arena;                                 // NTOK*512
    float* tmp   = xenc + (size_t)NTOK * 512;             // NTOK*512
    float* pfseq = tmp + (size_t)NTOK * 512;              // BJ*512
    float* qkv   = pfseq + (size_t)BJ * 512;              // NTOK*1536
    float* attno = qkv + (size_t)NTOK * 1536;             // NTOK*512
    float* ffn1  = qkv;                                   // NTOK*2048 (aliases qkv+attno, dead by then)

    launch_gemm(stream, paths, 512, p_emb_w, p_emb_b, xenc, NTOK, 512, 512, 0);
    launch_ew(NTOK * 512, g);
    add_pos_kernel<<<g, 256, 0, stream>>>(xenc, p_pos, 512);

    for (int l = 0; l < 2; ++l) {
      launch_gemm(stream, xenc, 512, pL[l].wqkv, pL[l].bqkv, qkv, NTOK, 512, 1536, 0);
      launch_ew(BJ * 4 * SEQLEN, g);
      attn_kernel<512, 4, false><<<g, 256, 0, stream>>>(qkv, nullptr, attno);
      launch_gemm(stream, attno, 512, pL[l].wo, pL[l].bo, tmp, NTOK, 512, 512, 0);
      add_ln_kernel<<<(NTOK + 7) / 8, 256, 0, stream>>>(xenc, tmp, pL[l].ln1_g, pL[l].ln1_b, 512, NTOK);
      launch_gemm(stream, xenc, 512, pL[l].w1, pL[l].b1, ffn1, NTOK, 512, 2048, 1);  // relu
      launch_gemm(stream, ffn1, 2048, pL[l].w2, pL[l].b2, tmp, NTOK, 2048, 512, 0);
      add_ln_kernel<<<(NTOK + 7) / 8, 256, 0, stream>>>(xenc, tmp, pL[l].ln2_g, pL[l].ln2_b, 512, NTOK);
    }
    // last token -> fc (strided A rows)
    launch_gemm(stream, xenc + 8 * 512, SEQLEN * 512, p_fc_w, p_fc_b, pfseq, BJ, 512, 512, 0);
    launch_ew(NGRAPH * 512, g);
    pf_mean_kernel<<<g, 256, 0, stream>>>(pfseq, pf);
  }

  // =========================================================================
  // Phase C: label transformer encoder (D=64, 4 heads, FFN=2048, masked)
  // =========================================================================
  {
    float* xl    = arena;                                 // NTOK*64
    float* tmpl  = xl + (size_t)NTOK * LDIM;              // NTOK*64
    float* lfseq = tmpl + (size_t)NTOK * LDIM;            // BJ*64
    float* qkvl  = lfseq + (size_t)BJ * LDIM;             // NTOK*192
    float* attnl = qkvl + (size_t)NTOK * 192;             // NTOK*64
    float* ffn1l = attnl + (size_t)NTOK * LDIM;           // NTOK*2048

    launch_ew(NTOK * LDIM, g);
    build_xl_kernel<<<g, 256, 0, stream>>>(labels, l_emb_w, l_emb_b, l_pos, xl);

    for (int l = 0; l < 2; ++l) {
      launch_gemm(stream, xl, LDIM, lL[l].wqkv, lL[l].bqkv, qkvl, NTOK, LDIM, 192, 0);
      launch_ew(BJ * 4 * SEQLEN, g);
      attn_kernel<64, 4, true><<<g, 256, 0, stream>>>(qkvl, masks, attnl);
      launch_gemm(stream, attnl, LDIM, lL[l].wo, lL[l].bo, tmpl, NTOK, LDIM, LDIM, 0);
      add_ln_kernel<<<(NTOK + 7) / 8, 256, 0, stream>>>(xl, tmpl, lL[l].ln1_g, lL[l].ln1_b, LDIM, NTOK);
      launch_gemm(stream, xl, LDIM, lL[l].w1, lL[l].b1, ffn1l, NTOK, LDIM, 2048, 1);
      launch_gemm(stream, ffn1l, 2048, lL[l].w2, lL[l].b2, tmpl, NTOK, 2048, LDIM, 0);
      add_ln_kernel<<<(NTOK + 7) / 8, 256, 0, stream>>>(xl, tmpl, lL[l].ln2_g, lL[l].ln2_b, LDIM, NTOK);
    }
    launch_gemm(stream, xl + 8 * LDIM, SEQLEN * LDIM, l_fc_w, l_fc_b, lfseq, BJ, LDIM, LDIM, 0);
    launch_ew(NGRAPH * NPATHS * LDIM, g);
    lf_build_kernel<<<g, 256, 0, stream>>>(lfseq, masks, lf);
  }

  // =========================================================================
  // Phase D: head MLP + output
  // =========================================================================
  {
    float* feat = arena;                                  // NGRAPH*FEATP (padded)
    float* hh1  = feat + (size_t)NGRAPH * FEATP;          // NGRAPH*512
    float* hh2  = hh1 + (size_t)NGRAPH * 512;             // NGRAPH*512

    launch_ew(NGRAPH * FEATP, g);
    feat_build_kernel<<<g, 256, 0, stream>>>(xsum1, xsum2, pf, lf, feat);
    launch_gemm(stream, feat, FEATP, w1p, h_b1, hh1, NGRAPH, FEATP, 512, 2);  // silu
    launch_gemm(stream, hh1, 512, h_w2, h_b2, hh2, NGRAPH, 512, 512, 2);      // silu
    launch_ew(NGRAPH, g);
    matvec_kernel<<<g, 256, 0, stream>>>(hh2, 512, h_w3, 1, 0, h_b3, xone, NGRAPH, 512);
    out_write_kernel<<<g, 256, 0, stream>>>(xone, xsum1, out);
  }

  (void)in_sizes; (void)n_in; (void)out_size; (void)ws_size;
}
